// behavior_model_39006892982744
// MI455X (gfx1250) — matrix operand order repaired and checked by reading, not compiled, not run on hardware
//
#include <hip/hip_runtime.h>
#include <cstdint>
#include <cstddef>

// ---------------------------------------------------------------------------
// Problem constants (from reference): B=128, N=32, C=1024, F=14, P=3, S=2
// K = B*N = 4096 boxes, tokens M = K*9 = 36864, conv-eff 1024->256,
// fc1 2304->1024 (+bias, relu), fc2 1024->27 (+bias).
// ---------------------------------------------------------------------------
#define NBOX   4096
#define CCH    1024
#define FMS    14
#define MTOK   36864     /* NBOX * 9 */
#define OEFF   256
#define FC1K   2304
#define FC1N   1024
#define FC2N   27
#define FC2NP  64        /* padded */

typedef _Float16 v16h __attribute__((ext_vector_type(16)));
typedef _Float16 v8h  __attribute__((ext_vector_type(8)));
typedef float    v8f  __attribute__((ext_vector_type(8)));

#define AS1 __attribute__((address_space(1)))
#define AS3 __attribute__((address_space(3)))

// ---------------------------------------------------------------------------
// Async global->LDS copy via CDNA5 GLOBAL_LOAD_ASYNC_TO_LDS_B128.
// The clang builtin on this toolchain has an incompatible signature, so emit
// the instruction directly: VDST = LDS byte address (AS3 offset), VADDR =
// 64-bit global address, saddr = off. Tracked by ASYNCcnt.
// ---------------------------------------------------------------------------
__device__ __forceinline__ void cp_async16(_Float16* lds_dst, const _Float16* gsrc) {
#if defined(__HIP_DEVICE_COMPILE__) && defined(__gfx1250__)
  unsigned ldsoff = (unsigned)(unsigned long long)(AS3 _Float16*)lds_dst;
  asm volatile("global_load_async_to_lds_b128 %0, %1, off"
               :: "v"(ldsoff), "v"(gsrc)
               : "memory");
#else
  *(v8h*)lds_dst = *(const v8h*)gsrc;
#endif
}

__device__ __forceinline__ void wait_async_copies() {
#if defined(__HIP_DEVICE_COMPILE__) && defined(__gfx1250__)
#if __has_builtin(__builtin_amdgcn_s_wait_asynccnt)
  __builtin_amdgcn_s_wait_asynccnt(0);
#else
  asm volatile("s_wait_asynccnt 0" ::: "memory");
#endif
#endif
}

__device__ __forceinline__ v16h combine8(v8h lo, v8h hi) {
  return __builtin_shufflevector(lo, hi, 0, 1, 2, 3, 4, 5, 6, 7,
                                 8, 9, 10, 11, 12, 13, 14, 15);
}

// ---------------------------------------------------------------------------
// Prep kernels: effective conv weight (w_c2 @ w_c1, stored transposed as
// [c][o] f16), f16 copy of w_fc1, padded f16 copy of w_fc2.
// ---------------------------------------------------------------------------
__global__ void k_weff(const float* __restrict__ w_c1,   // [512][1024]
                       const float* __restrict__ w_c2,   // [256][512]
                       _Float16* __restrict__ w_effT) {  // [1024][256]
  int idx = blockIdx.x * 256 + threadIdx.x;              // 262144 total
  int o = idx >> 10;
  int c = idx & 1023;
  float s = 0.f;
  for (int m = 0; m < 512; ++m)
    s += w_c2[o * 512 + m] * w_c1[m * 1024 + c];
  w_effT[(size_t)c * OEFF + o] = (_Float16)s;
}

__global__ void k_cvt_f16(const float* __restrict__ src, _Float16* __restrict__ dst, int n) {
  int idx = blockIdx.x * 256 + threadIdx.x;
  if (idx < n) dst[idx] = (_Float16)src[idx];
}

__global__ void k_wfc2_pad(const float* __restrict__ w_fc2,  // [1024][27]
                           _Float16* __restrict__ dst) {     // [1024][64]
  int idx = blockIdx.x * 256 + threadIdx.x;                  // 65536
  int k = idx >> 6, n = idx & 63;
  dst[idx] = (n < FC2N) ? (_Float16)w_fc2[k * FC2N + n] : (_Float16)0.f;
}

// ---------------------------------------------------------------------------
// ROIAlign (torchvision bilinear semantics) -> pooled f16 [token][c],
// token = box*9 + (py*3+px). One block per box, 256 threads over channels.
// fmap (103 MB) is L2-resident on MI455X (192 MB L2).
// ---------------------------------------------------------------------------
__global__ void k_roi(const float* __restrict__ fmap,    // [128][1024][14][14]
                      const float* __restrict__ boxes,   // [128][32][4]
                      _Float16* __restrict__ pooled) {   // [36864][1024]
  const int kbox = blockIdx.x;
  const int img = kbox >> 5;
  const float* bx = boxes + (size_t)kbox * 4;
  float x1 = fminf(fmaxf(bx[0], 0.f), (float)FMS);
  float y1 = fminf(fmaxf(bx[1], 0.f), (float)FMS);
  float x2 = fminf(fmaxf(bx[2], 0.f), (float)FMS);
  float y2 = fminf(fmaxf(bx[3], 0.f), (float)FMS);
  float bw = fmaxf(x2 - x1, 1.f) * (1.f / 3.f);
  float bh = fmaxf(y2 - y1, 1.f) * (1.f / 3.f);

  int ylo[6], xlo[6];
  float wy0[6], wy1[6], wx0[6], wx1[6];
#pragma unroll
  for (int j = 0; j < 6; ++j) {
    float pos = (float)(j >> 1) + ((float)(j & 1) + 0.5f) * 0.5f;  // bin + off
    // y axis
    float yc = y1 + pos * bh;
    bool vy = (yc > -1.f) && (yc < (float)FMS);
    float cy = fminf(fmaxf(yc, 0.f), (float)(FMS - 1));
    int ly = (int)floorf(cy); ly = ly < 0 ? 0 : (ly > FMS - 2 ? FMS - 2 : ly);
    float fy = cy - (float)ly;
    ylo[j] = ly; wy0[j] = vy ? 1.f - fy : 0.f; wy1[j] = vy ? fy : 0.f;
    // x axis
    float xc = x1 + pos * bw;
    bool vx = (xc > -1.f) && (xc < (float)FMS);
    float cx = fminf(fmaxf(xc, 0.f), (float)(FMS - 1));
    int lx = (int)floorf(cx); lx = lx < 0 ? 0 : (lx > FMS - 2 ? FMS - 2 : lx);
    float fx = cx - (float)lx;
    xlo[j] = lx; wx0[j] = vx ? 1.f - fx : 0.f; wx1[j] = vx ? fx : 0.f;
  }

  const float* fbase = fmap + (size_t)img * CCH * (FMS * FMS);
  _Float16* prow = pooled + (size_t)kbox * 9 * CCH;
  for (int c = threadIdx.x; c < CCH; c += 256) {
    const float* fc = fbase + (size_t)c * (FMS * FMS);
    float acc9[9];
#pragma unroll
    for (int q = 0; q < 9; ++q) acc9[q] = 0.f;
#pragma unroll
    for (int j = 0; j < 6; ++j) {
      const float* r0 = fc + ylo[j] * FMS;
      const float* r1 = r0 + FMS;
      float w0 = wy0[j], w1 = wy1[j];
#pragma unroll
      for (int i = 0; i < 6; ++i) {
        float v0 = r0[xlo[i]] * wx0[i] + r0[xlo[i] + 1] * wx1[i];
        float v1 = r1[xlo[i]] * wx0[i] + r1[xlo[i] + 1] * wx1[i];
        acc9[(j >> 1) * 3 + (i >> 1)] += v0 * w0 + v1 * w1;
      }
    }
#pragma unroll
    for (int q = 0; q < 9; ++q)
      prow[(size_t)q * CCH + c] = (_Float16)(acc9[q] * 0.25f);
  }
}

// ---------------------------------------------------------------------------
// WMMA GEMM core: C[64x64] += A[64xK] * B[Kx64], f16 in / f32 accum.
// 128 threads = 4 waves, each wave owns a 32x32 quadrant = 2x2 WMMA tiles.
// A staged to LDS row-major (async global->LDS b128), B staged transposed
// so B fragments are contiguous ds_load_b128 reads.
// Fragment layouts per CDNA5 ISA 7.12.2 (wave32).
// ---------------------------------------------------------------------------
__device__ __forceinline__ void gemm_tile_core(
    const _Float16* __restrict__ A, int lda,
    const _Float16* __restrict__ B, int ldb,
    int K, int gm0, int gn0, v8f acc[2][2]) {
  __shared__ __align__(16) _Float16 As[64][40];  // [m][k], pad 40 halfs
  __shared__ __align__(16) _Float16 Bs[64][40];  // [n][k] (transposed)

  const int tid = threadIdx.x;
  const int lane = tid & 31;
  const int wave = tid >> 5;
  const int wm = (wave >> 1) * 32;
  const int wn = (wave & 1) * 32;
  const int l15 = lane & 15;
  const int kbA = (lane < 16) ? 0 : 8;
  const int kbB = (lane < 16) ? 0 : 8;

  for (int k0 = 0; k0 < K; k0 += 32) {
    __syncthreads();
    // Stage A: 64 rows x 32 halfs = 256 x 16B segments; 2 per thread (async)
#pragma unroll
    for (int e = 0; e < 2; ++e) {
      int si = tid + e * 128;
      int row = si >> 2, sub = si & 3;
      cp_async16(&As[row][sub * 8],
                 A + (size_t)(gm0 + row) * lda + k0 + sub * 8);
    }
    // Stage B transposed: 32 k-rows x 64 cols; read v8h along n, scatter to [n][k]
#pragma unroll
    for (int e = 0; e < 2; ++e) {
      int si = tid + e * 128;
      int kk = si >> 3, ns = si & 7;
      v8h gb = *(const v8h*)(B + (size_t)(k0 + kk) * ldb + gn0 + ns * 8);
#pragma unroll
      for (int j = 0; j < 8; ++j) Bs[ns * 8 + j][kk] = gb[j];
    }
    wait_async_copies();
    __syncthreads();

    v16h af[2], bf[2];
#pragma unroll
    for (int r = 0; r < 2; ++r) {
      int m = wm + r * 16 + l15;
      af[r] = combine8(*(const v8h*)&As[m][kbA], *(const v8h*)&As[m][16 + kbA]);
    }
#pragma unroll
    for (int s = 0; s < 2; ++s) {
      int n = wn + s * 16 + l15;
      bf[s] = combine8(*(const v8h*)&Bs[n][kbB], *(const v8h*)&Bs[n][16 + kbB]);
    }
#pragma unroll
    for (int r = 0; r < 2; ++r)
#pragma unroll
      for (int s = 0; s < 2; ++s)
        acc[r][s] = __builtin_amdgcn_wmma_f32_16x16x32_f16(
            false, af[r], false, bf[s], (short)0, acc[r][s], false, false);
  }
}

__device__ __forceinline__ void zero_acc(v8f acc[2][2]) {
#pragma unroll
  for (int r = 0; r < 2; ++r)
#pragma unroll
    for (int s = 0; s < 2; ++s)
#pragma unroll
      for (int i = 0; i < 8; ++i) acc[r][s][i] = 0.f;
}

// GEMM 1: x2[t,o] = pooled[t,:] . w_effT[:,o]; write into fc1 layout
// y[k][o*9+hw] with t = k*9+hw.   M=36864, N=256, K=1024.
__global__ void k_conv_gemm(const _Float16* __restrict__ A,   // pooled [36864][1024]
                            const _Float16* __restrict__ B,   // w_effT [1024][256]
                            _Float16* __restrict__ Y) {       // [4096][2304]
  v8f acc[2][2];
  zero_acc(acc);
  const int gm0 = blockIdx.x * 64, gn0 = blockIdx.y * 64;
  gemm_tile_core(A, CCH, B, OEFF, CCH, gm0, gn0, acc);

  const int lane = threadIdx.x & 31, wave = threadIdx.x >> 5;
  const int wm = (wave >> 1) * 32, wn = (wave & 1) * 32;
  const int l15 = lane & 15, mb = (lane < 16) ? 0 : 8;
#pragma unroll
  for (int r = 0; r < 2; ++r)
#pragma unroll
    for (int s = 0; s < 2; ++s) {
      int o = gn0 + wn + s * 16 + l15;
#pragma unroll
      for (int i = 0; i < 8; ++i) {
        int t = gm0 + wm + r * 16 + mb + i;
        int k = t / 9, hw = t - k * 9;
        Y[(size_t)k * FC1K + o * 9 + hw] = (_Float16)acc[r][s][i];
      }
    }
}

// GEMM 2: h = relu(Y @ w_fc1 + b_fc1).  M=4096, N=1024, K=2304.
__global__ void k_fc1_gemm(const _Float16* __restrict__ A,   // Y [4096][2304]
                           const _Float16* __restrict__ B,   // w_fc1 [2304][1024]
                           const float* __restrict__ bias,   // [1024]
                           _Float16* __restrict__ H) {       // [4096][1024]
  v8f acc[2][2];
  zero_acc(acc);
  const int gm0 = blockIdx.x * 64, gn0 = blockIdx.y * 64;
  gemm_tile_core(A, FC1K, B, FC1N, FC1K, gm0, gn0, acc);

  const int lane = threadIdx.x & 31, wave = threadIdx.x >> 5;
  const int wm = (wave >> 1) * 32, wn = (wave & 1) * 32;
  const int l15 = lane & 15, mb = (lane < 16) ? 0 : 8;
#pragma unroll
  for (int r = 0; r < 2; ++r)
#pragma unroll
    for (int s = 0; s < 2; ++s) {
      int n = gn0 + wn + s * 16 + l15;
      float bn = bias[n];
#pragma unroll
      for (int i = 0; i < 8; ++i) {
        int m = gm0 + wm + r * 16 + mb + i;
        float v = acc[r][s][i] + bn;
        H[(size_t)m * FC1N + n] = (_Float16)(v > 0.f ? v : 0.f);
      }
    }
}

// GEMM 3: logits = H @ w_fc2 + b_fc2.  M=4096, N=64(pad, 27 valid), K=1024.
__global__ void k_fc2_gemm(const _Float16* __restrict__ A,   // H [4096][1024]
                           const _Float16* __restrict__ B,   // w_fc2p [1024][64]
                           const float* __restrict__ bias,   // [27]
                           float* __restrict__ out) {        // [4096][27]
  v8f acc[2][2];
  zero_acc(acc);
  const int gm0 = blockIdx.x * 64;
  gemm_tile_core(A, FC1N, B, FC2NP, FC1N, gm0, 0, acc);

  const int lane = threadIdx.x & 31, wave = threadIdx.x >> 5;
  const int wm = (wave >> 1) * 32, wn = (wave & 1) * 32;
  const int l15 = lane & 15, mb = (lane < 16) ? 0 : 8;
#pragma unroll
  for (int r = 0; r < 2; ++r)
#pragma unroll
    for (int s = 0; s < 2; ++s) {
      int n = wn + s * 16 + l15;
      if (n < FC2N) {
        float bn = bias[n];
#pragma unroll
        for (int i = 0; i < 8; ++i) {
          int m = gm0 + wm + r * 16 + mb + i;
          out[(size_t)m * FC2N + n] = acc[r][s][i] + bn;
        }
      }
    }
}

// ---------------------------------------------------------------------------
// Host launcher. Workspace layout (~108 MB, all 256B aligned):
//   w_effT f16 [1024][256], w_fc1 f16 [2304][1024], w_fc2p f16 [1024][64],
//   pooled f16 [36864][1024], Y f16 [4096][2304], H f16 [4096][1024].
// ---------------------------------------------------------------------------
extern "C" void kernel_launch(void* const* d_in, const int* in_sizes, int n_in,
                              void* d_out, int out_size, void* d_ws, size_t ws_size,
                              hipStream_t stream) {
  (void)in_sizes; (void)n_in; (void)out_size; (void)ws_size;
  const float* fmap  = (const float*)d_in[0];
  const float* boxes = (const float*)d_in[1];
  const float* w_c1  = (const float*)d_in[2];
  const float* w_c2  = (const float*)d_in[3];
  const float* w_fc1 = (const float*)d_in[4];
  const float* b_fc1 = (const float*)d_in[5];
  const float* w_fc2 = (const float*)d_in[6];
  const float* b_fc2 = (const float*)d_in[7];
  float* out = (float*)d_out;

  char* ws = (char*)d_ws;
  _Float16* w_effT = (_Float16*)(ws + 0);                 // 524288 B
  _Float16* wfc1h  = (_Float16*)(ws + 524288);            // 4718592 B
  _Float16* wfc2h  = (_Float16*)(ws + 5243136);           // 131072 B
  _Float16* pooled = (_Float16*)(ws + 5374208);           // 75497472 B
  _Float16* ybuf   = (_Float16*)(ws + 80871680);          // 18874368 B
  _Float16* hbuf   = (_Float16*)(ws + 99746048);          // 8388608 B -> end ~108.1 MB

  // weight prep
  k_weff<<<1024, 256, 0, stream>>>(w_c1, w_c2, w_effT);
  k_cvt_f16<<<(FC1K * FC1N + 255) / 256, 256, 0, stream>>>(w_fc1, wfc1h, FC1K * FC1N);
  k_wfc2_pad<<<(1024 * FC2NP) / 256, 256, 0, stream>>>(w_fc2, wfc2h);

  // ROIAlign -> pooled f16 tokens
  k_roi<<<NBOX, 256, 0, stream>>>(fmap, boxes, pooled);

  // fused conv1+conv2 GEMM: [36864 x 1024] x [1024 x 256]
  k_conv_gemm<<<dim3(MTOK / 64, OEFF / 64), 128, 0, stream>>>(pooled, w_effT, ybuf);

  // fc1 GEMM + bias + relu: [4096 x 2304] x [2304 x 1024]
  k_fc1_gemm<<<dim3(NBOX / 64, FC1N / 64), 128, 0, stream>>>(ybuf, wfc1h, b_fc1, hbuf);

  // fc2 GEMM + bias -> logits f32: [4096 x 1024] x [1024 x 64(pad 27)]
  k_fc2_gemm<<<dim3(NBOX / 64, 1), 128, 0, stream>>>(hbuf, wfc2h, b_fc2, out);
}
